// RGATLayer_55533927137534
// MI455X (gfx1250) — matrix operand order repaired and checked by reading, not compiled, not run on hardware
//
#include <hip/hip_runtime.h>

typedef __attribute__((ext_vector_type(16))) __bf16 v16bf;
typedef __attribute__((ext_vector_type(8)))  float  v8f;
typedef __attribute__((ext_vector_type(4)))  float  v4f;

#define NP 100000
#define NA 60000
#define NS 1000
#define NTOT (NP + NA + NS)
#define H 8
#define DH 16
#define F 128

// ---------------------------------------------------------------------------
// Dense transform: C[M,128] = A[M,128] @ W[128,128], f32 in, bf16 WMMA, f32 out.
// Fused epilogue: el[n,h] = sum_d C[n, h*16+d] * al[h*16+d]   (tile tn == head h)
// Block = 256 threads = 8 waves; each wave computes 16 rows x 128 cols.
// C tile is staged through LDS so global stores are 2x b128 per lane.
// ---------------------------------------------------------------------------
__global__ __launch_bounds__(256) void rgat_gemm_bf16(
    const float* __restrict__ A, const float* __restrict__ W,
    const float* __restrict__ al, float* __restrict__ C,
    float* __restrict__ el, int M)
{
    __shared__ __bf16 ldsWT[F * F];                  // W^T as bf16 (32 KB)
    __shared__ float  ldsAl[F];
    __shared__ __align__(16) float ldsC[8 * 16 * 20]; // per-wave 16 rows, stride 20 (10 KB)

    const int t = threadIdx.x;
    // Cooperative load of W, transpose + convert to bf16 in LDS.
    for (int idx = t; idx < F * F; idx += 256) {
        int k = idx >> 7, n = idx & 127;
        ldsWT[n * F + k] = (__bf16)W[idx];
    }
    if (t < F) ldsAl[t] = al[t];
    __syncthreads();

    const int wave = t >> 5;
    const int lane = t & 31;
    const int half = lane >> 4;     // ISA A/B fragment half selector
    const int lrow = lane & 15;
    const int row0 = blockIdx.x * 128 + wave * 16;
    const bool fullTile = (row0 + 16 <= M);   // wave-uniform

    int arow_i = row0 + lrow;
    if (arow_i >= M) arow_i = M - 1;          // clamp loads; stores guarded
    const float* arow = A + (size_t)arow_i * F;

    // Build 4 A fragments (K chunks of 32). contiguous K order (same as B):
    // elements 0..7  <-> K = 32c + 16*half + j
    // elements 8..15 <-> K = 32c + 16*half + 8 + j
    v16bf afrag[4];
#pragma unroll
    for (int c = 0; c < 4; ++c) {
        const int k0 = 32 * c + 16 * half;
        const int k1 = k0 + 8;
#pragma unroll
        for (int j = 0; j < 8; ++j) afrag[c][j] = (__bf16)arow[k0 + j];
#pragma unroll
        for (int j = 0; j < 8; ++j) afrag[c][8 + j] = (__bf16)arow[k1 + j];
    }

    float* myC = ldsC + wave * (16 * 20);
    const int rr = lane >> 1;        // staged-read row (0..15)
    const int c0 = (lane & 1) * 8;   // staged-read col base {0,8}
    const int r  = row0 + rr;        // global row for stores

#pragma unroll
    for (int tn = 0; tn < 8; ++tn) {
        v8f acc = {};
#pragma unroll
        for (int c = 0; c < 4; ++c) {
            // B frag: W^T row n, K = 32c + 16*half .. +15 (contiguous in LDS)
            const v16bf bfrag =
                *(const v16bf*)(&ldsWT[(tn * 16 + lrow) * F + 32 * c + 16 * half]);
            acc = __builtin_amdgcn_wmma_f32_16x16x32_bf16(
                false, afrag[c], false, bfrag, (short)0, acc, false, false);
        }

        // Stage tile: lane holds col lrow; VGPR v -> tile row 8*half + v.
        // Stride 20 => write banks disjoint between lane halves.
#pragma unroll
        for (int v = 0; v < 8; ++v)
            myC[(8 * half + v) * 20 + lrow] = acc[v];

        // Read back row-contiguous (LDS ops from one wave are in-order; the
        // compiler inserts the dscnt wait for the RAW).
        v4f lo = *(const v4f*)&myC[rr * 20 + c0];
        v4f hi = *(const v4f*)&myC[rr * 20 + c0 + 4];

        // Fused el partial: this tile is head h == tn.
        float psum = 0.f;
#pragma unroll
        for (int j = 0; j < 4; ++j) psum += lo[j] * ldsAl[tn * 16 + c0 + j];
#pragma unroll
        for (int j = 0; j < 4; ++j) psum += hi[j] * ldsAl[tn * 16 + c0 + 4 + j];
        psum += __shfl_xor(psum, 1, 32);   // combine the two half-rows

        float* dst = &C[(size_t)r * F + tn * 16 + c0];
        if (fullTile) {
            *(v4f*)dst       = lo;
            *(v4f*)(dst + 4) = hi;
            if ((lane & 1) == 0) el[(size_t)r * H + tn] = psum;
        } else if (r < M) {
            *(v4f*)dst       = lo;
            *(v4f*)(dst + 4) = hi;
            if ((lane & 1) == 0) el[(size_t)r * H + tn] = psum;
        }
    }
}

// Vr[k,h] = sum_d W[k, h*16+d] * ar[h*16+d]   (128x8)
__global__ void rgat_vr(const float* __restrict__ W, const float* __restrict__ ar,
                        float* __restrict__ Vr)
{
    int k = threadIdx.x;  // 128 threads
#pragma unroll
    for (int h = 0; h < H; ++h) {
        float s = 0.f;
#pragma unroll
        for (int d = 0; d < DH; ++d) s += W[k * F + h * DH + d] * ar[h * DH + d];
        Vr[k * H + h] = s;
    }
}

// er[n,h] = feat[n,:] . Vr[:,h]
__global__ void rgat_er(const float* __restrict__ feat, const float* __restrict__ Vr,
                        float* __restrict__ er, int N)
{
    int i = blockIdx.x * blockDim.x + threadIdx.x;
    if (i >= N * H) return;
    int n = i >> 3, h = i & 7;
    const float* row = feat + (size_t)n * F;
    float s = 0.f;
#pragma unroll 16
    for (int k = 0; k < F; ++k) s += row[k] * Vr[k * H + h];
    er[i] = s;
}

__global__ void rgat_init_seg(unsigned* __restrict__ emax, float* __restrict__ denom, int n)
{
    int i = blockIdx.x * blockDim.x + threadIdx.x;
    if (i < n) { emax[i] = 0u; denom[i] = 0.f; }
}

__device__ __forceinline__ unsigned fkey(float f) {
    unsigned u = __float_as_uint(f);
    return (u & 0x80000000u) ? ~u : (u | 0x80000000u);
}
__device__ __forceinline__ float funkey(unsigned k) {
    return __uint_as_float((k & 0x80000000u) ? (k ^ 0x80000000u) : ~k);
}
__device__ __forceinline__ float lrelu(float x) { return x > 0.f ? x : 0.2f * x; }

__global__ void rgat_edge_max(const int* __restrict__ src, const int* __restrict__ dst,
                              const float* __restrict__ el, const float* __restrict__ er,
                              unsigned* __restrict__ emax, int E)
{
    int i = blockIdx.x * blockDim.x + threadIdx.x;
    if (i >= E * H) return;
    int e = i >> 3, h = i & 7;
    int s = src[e], d = dst[e];
    float x = lrelu(el[s * H + h] + er[d * H + h]);
    atomicMax(&emax[d * H + h], fkey(x));
}

__global__ void rgat_edge_sum(const int* __restrict__ src, const int* __restrict__ dst,
                              const float* __restrict__ el, const float* __restrict__ er,
                              const unsigned* __restrict__ emax, float* __restrict__ denom,
                              int E)
{
    int i = blockIdx.x * blockDim.x + threadIdx.x;
    if (i >= E * H) return;
    int e = i >> 3, h = i & 7;
    int s = src[e], d = dst[e];
    float x = lrelu(el[s * H + h] + er[d * H + h]);
    float m = funkey(emax[d * H + h]);
    atomicAdd(&denom[d * H + h], expf(x - m));
}

// One thread per (edge, feature-component): alpha * hs[src] scattered into out.
__global__ void rgat_edge_aggr(const int* __restrict__ src, const int* __restrict__ dst,
                               const float* __restrict__ el, const float* __restrict__ er,
                               const unsigned* __restrict__ emax, const float* __restrict__ denom,
                               const float* __restrict__ hs, float* __restrict__ out, int E)
{
    int i = blockIdx.x * blockDim.x + threadIdx.x;
    if (i >= E * F) return;
    int e = i >> 7, c = i & 127, h = c >> 4;
    int s = src[e], d = dst[e];
    float x = lrelu(el[s * H + h] + er[d * H + h]);
    float m = funkey(emax[d * H + h]);
    float alpha = expf(x - m) / fmaxf(denom[d * H + h], 1e-9f);
    atomicAdd(&out[(size_t)d * F + c], alpha * hs[(size_t)s * F + c]);
}

__global__ void rgat_zero(float* __restrict__ p, int n)
{
    int i = blockIdx.x * blockDim.x + threadIdx.x;
    if (i < n) p[i] = 0.f;
}

// Paper: out = 0.5*out + 0.5*(b1+b3); Author: += b0; Subject: += b2
__global__ void rgat_finalize(float* __restrict__ out, const float* __restrict__ bs, int n)
{
    int i = blockIdx.x * blockDim.x + threadIdx.x;
    if (i >= n) return;
    int c = i & 127;
    int node = (i >> 7) % NTOT;
    if (node < NP)            out[i] = out[i] * 0.5f + 0.5f * (bs[1 * F + c] + bs[3 * F + c]);
    else if (node < NP + NA)  out[i] += bs[0 * F + c];
    else                      out[i] += bs[2 * F + c];
}

static inline int cdiv(int a, int b) { return (a + b - 1) / b; }

extern "C" void kernel_launch(void* const* d_in, const int* in_sizes, int n_in,
                              void* d_out, int out_size, void* d_ws, size_t ws_size,
                              hipStream_t stream)
{
    const float* feats[2][3] = {
        {(const float*)d_in[0], (const float*)d_in[1], (const float*)d_in[2]},
        {(const float*)d_in[3], (const float*)d_in[4], (const float*)d_in[5]},
    };
    const float* Ws  = (const float*)d_in[14];  // [4,128,128]
    const float* als = (const float*)d_in[15];  // [4,8,16]
    const float* ars = (const float*)d_in[16];  // [4,8,16]
    const float* bs  = (const float*)d_in[17];  // [4,128]
    float* out = (float*)d_out;

    // workspace layout (all offsets 256B aligned)
    char* ws = (char*)d_ws;
    float*    hs    = (float*)   (ws);                         // 100000*128*4 = 51,200,000
    float*    el    = (float*)   (ws + 51200000);              //   3,200,000
    float*    er    = (float*)   (ws + 51200000 + 3200000);    //   3,200,000
    unsigned* emax  = (unsigned*)(ws + 51200000 + 6400000);    //   3,200,000
    float*    denom = (float*)   (ws + 51200000 + 9600000);    //   3,200,000
    float*    Vr    = (float*)   (ws + 51200000 + 12800000);   //       4,096

    // zero output accumulators
    rgat_zero<<<cdiv(out_size, 256), 256, 0, stream>>>(out, out_size);

    const int ntype[3] = {NP, NA, NS};
    // relation table: srcType, dstType, src-edge input idx, dst-edge input idx, out offset
    const int relSrcT[4] = {0, 1, 0, 2};
    const int relDstT[4] = {1, 0, 2, 0};
    const int relSEI[4]  = {6, 8, 10, 12};
    const int relDEI[4]  = {7, 9, 11, 13};
    const size_t relOut[4] = {(size_t)NP * F, 0, (size_t)(NP + NA) * F, 0};

    for (int inp = 0; inp < 2; ++inp) {
        float* outBase = out + (size_t)inp * NTOT * F;
        for (int r = 0; r < 4; ++r) {
            const float* srcF = feats[inp][relSrcT[r]];
            const float* dstF = feats[inp][relDstT[r]];
            const int Ns = ntype[relSrcT[r]];
            const int Nd = ntype[relDstT[r]];
            const int* esrc = (const int*)d_in[relSEI[r]];
            const int* edst = (const int*)d_in[relDEI[r]];
            const int E = in_sizes[relSEI[r]];
            const float* Wr  = Ws  + (size_t)r * F * F;
            const float* alr = als + (size_t)r * F;
            const float* arr = ars + (size_t)r * F;
            float* outSlice = outBase + relOut[r];

            // 1) hs = srcF @ Wr (bf16 WMMA) with fused el epilogue
            rgat_gemm_bf16<<<cdiv(Ns, 128), 256, 0, stream>>>(srcF, Wr, alr, hs, el, Ns);
            // 2) Vr = Wr . ar ;  er = dstF @ Vr
            rgat_vr<<<1, 128, 0, stream>>>(Wr, arr, Vr);
            rgat_er<<<cdiv(Nd * H, 256), 256, 0, stream>>>(dstF, Vr, er, Nd);
            // 3) segment softmax state init
            rgat_init_seg<<<cdiv(Nd * H, 256), 256, 0, stream>>>(emax, denom, Nd * H);
            // 4) max, 5) denom, 6) aggregate
            rgat_edge_max <<<cdiv(E * H, 256), 256, 0, stream>>>(esrc, edst, el, er, emax, E);
            rgat_edge_sum <<<cdiv(E * H, 256), 256, 0, stream>>>(esrc, edst, el, er, emax, denom, E);
            rgat_edge_aggr<<<cdiv(E * F, 256), 256, 0, stream>>>(esrc, edst, el, er, emax, denom,
                                                                 hs, outSlice, E);
        }
    }
    // bias + paper mean
    rgat_finalize<<<cdiv(out_size, 256), 256, 0, stream>>>(out, bs, out_size);
}
